// TreeStructuredHierarchicalPrototypes_6837587935765
// MI455X (gfx1250) — matrix operand order repaired and checked by reading, not compiled, not run on hardware
//
#include <hip/hip_runtime.h>

// ---------------------------------------------------------------------------
// Problem constants (from the reference)
// ---------------------------------------------------------------------------
#define NN 65536   // n_support
#define DD 512     // embed dim
#define HH 256     // hidden dim of cluster head
#define KC 16      // max clusters
#define LL 3       // levels
#define CC 64      // classes
#define CK (CC * KC)

typedef __bf16 bf16;
typedef bf16 v16bf __attribute__((ext_vector_type(16)));
typedef float v8f __attribute__((ext_vector_type(8)));
typedef unsigned int q4 __attribute__((ext_vector_type(4)));

union Frag { v16bf v; q4 u[2]; };

// D = A(16x32 bf16) x B(32x16 bf16) + C(16x16 f32), wave32.
static __device__ __forceinline__ v8f wmma_bf16(const Frag& a, const Frag& b, v8f c) {
  return __builtin_amdgcn_wmma_f32_16x16x32_bf16(false, a.v, false, b.v, (short)0, c,
                                                 false, false);
}

// Async global->LDS copy of one 16B chunk per lane (GV mode, ASYNCcnt-tracked).
static __device__ __forceinline__ void async_copy_b128(unsigned int ldsByteAddr,
                                                       const void* gptr) {
  asm volatile("global_load_async_to_lds_b128 %0, %1, off"
               :
               : "v"(ldsByteAddr), "v"((unsigned long long)(uintptr_t)gptr)
               : "memory");
}
static __device__ __forceinline__ void wait_asynccnt0() {
  asm volatile("s_wait_asynccnt 0" ::: "memory");
}

// ---------------------------------------------------------------------------
// Elementwise helpers
// ---------------------------------------------------------------------------
__global__ void init_current_kernel(const float* __restrict__ sf,
                                    float* __restrict__ curF,
                                    bf16* __restrict__ curB, int total) {
  int i = blockIdx.x * blockDim.x + threadIdx.x;
  if (i >= total) return;
  float v = sf[i];
  curF[i] = v;
  curB[i] = (bf16)v;
}

__global__ void zero_f32_kernel(float* __restrict__ p, int n) {
  int i = blockIdx.x * blockDim.x + threadIdx.x;
  if (i < n) p[i] = 0.0f;
}

// W [Kdim, Nn] (row-major) -> WT [Nn, Kdim] bf16  (B operand layout for WMMA)
__global__ void transpose_bf16_kernel(const float* __restrict__ W,
                                      bf16* __restrict__ WT, int Kdim, int Nn) {
  int idx = blockIdx.x * blockDim.x + threadIdx.x;
  if (idx >= Kdim * Nn) return;
  int n = idx / Kdim;
  int k = idx - n * Kdim;
  WT[idx] = (bf16)W[k * Nn + n];
}

__global__ void finalize_means_kernel(const float* __restrict__ sums,
                                      const float* __restrict__ cnt,
                                      bf16* __restrict__ meansB, int total) {
  int idx = blockIdx.x * blockDim.x + threadIdx.x;
  if (idx >= total) return;
  int s = idx >> 9;  // / DD
  meansB[idx] = (bf16)(sums[idx] / fmaxf(cnt[s], 1.0f));
}

// one wave per row; 512 cols, 16 per lane
__global__ __launch_bounds__(32) void layernorm_relu_kernel(
    const float* __restrict__ X, const float* __restrict__ g,
    const float* __restrict__ b, bf16* __restrict__ Y) {
  int row = blockIdx.x;
  int lane = threadIdx.x;
  const float* x = X + (size_t)row * DD + lane * 16;
  float v[16];
  float s = 0.0f;
#pragma unroll
  for (int j = 0; j < 16; ++j) { v[j] = x[j]; s += v[j]; }
#pragma unroll
  for (int off = 16; off >= 1; off >>= 1) s += __shfl_xor(s, off);
  float m = s * (1.0f / (float)DD);
  float s2 = 0.0f;
#pragma unroll
  for (int j = 0; j < 16; ++j) { float d = v[j] - m; s2 += d * d; }
#pragma unroll
  for (int off = 16; off >= 1; off >>= 1) s2 += __shfl_xor(s2, off);
  float r = rsqrtf(s2 * (1.0f / (float)DD) + 1e-5f);
#pragma unroll
  for (int j = 0; j < 16; ++j) {
    int col = lane * 16 + j;
    float y = (v[j] - m) * r * g[col] + b[col];
    Y[(size_t)row * DD + col] = (bf16)fmaxf(y, 0.0f);
  }
}

// masked mean over non-empty clusters -> lp [C,D] f32 + combined [C, L*D] bf16
__global__ void level_proto_kernel(const float* __restrict__ refined,
                                   const float* __restrict__ cnt,
                                   float* __restrict__ lpF,
                                   bf16* __restrict__ combB, int level) {
  int idx = blockIdx.x * blockDim.x + threadIdx.x;
  if (idx >= CC * DD) return;
  int c = idx >> 9;
  int d = idx & (DD - 1);
  float s = 0.0f, ne = 0.0f;
#pragma unroll
  for (int k = 0; k < KC; ++k) {
    float nz = (cnt[c * KC + k] > 0.0f) ? 1.0f : 0.0f;
    ne += nz;
    s += refined[(size_t)(c * KC + k) * DD + d] * nz;
  }
  float v = s / fmaxf(ne, 1.0f);
  lpF[idx] = v;
  combB[(size_t)c * (LL * DD) + level * DD + d] = (bf16)v;
}

__global__ void update_current_kernel(float* __restrict__ curF,
                                      bf16* __restrict__ curB,
                                      const float* __restrict__ lpF,
                                      const int* __restrict__ labels, int total) {
  int idx = blockIdx.x * blockDim.x + threadIdx.x;
  if (idx >= total) return;
  int i = idx >> 9;
  int d = idx & (DD - 1);
  float v = curF[idx] + lpF[labels[i] * DD + d];
  curF[idx] = v;
  curB[idx] = (bf16)v;
}

// ---------------------------------------------------------------------------
// Generic WMMA GEMM:  C[M,Nn] = A[M,Kdim](bf16) * BT[Nn,Kdim](bf16)^T + bias
// one wave per 16x16 output tile.  Optional ReLU, optional f32/bf16 outputs.
// ---------------------------------------------------------------------------
__global__ __launch_bounds__(32) void wmma_gemm_kernel(
    const bf16* __restrict__ A, const bf16* __restrict__ BT,
    const float* __restrict__ bias, float* __restrict__ Cf,
    bf16* __restrict__ Cb, int M, int Nn, int Kdim, int relu) {
  int lane = threadIdx.x & 31;
  int half = lane >> 4;
  int l16 = lane & 15;
  int ntN = Nn >> 4;
  int tM = (blockIdx.x / ntN) << 4;
  int tN = (blockIdx.x % ntN) << 4;
  const bf16* aRow = A + (size_t)(tM + l16) * Kdim;
  const bf16* bRow = BT + (size_t)(tN + l16) * Kdim;
  v8f acc = {};
  for (int k0 = 0; k0 < Kdim; k0 += 32) {
    __builtin_prefetch(aRow + k0 + 128, 0, 0);  // global_prefetch_b8
    Frag a, b;
    int ca = k0 + half * 8;   // A: lane-half K chunks {ca..ca+7, ca+16..ca+23}
    a.u[0] = *(const q4*)(aRow + ca);
    a.u[1] = *(const q4*)(aRow + ca + 16);
    int cb = k0 + half * 8;   // B: lane-half K chunks {cb..cb+7, cb+16..cb+23}
    b.u[0] = *(const q4*)(bRow + cb);
    b.u[1] = *(const q4*)(bRow + cb + 16);
    acc = wmma_bf16(a, b, acc);
  }
  float bv = bias ? bias[tN + l16] : 0.0f;
#pragma unroll
  for (int r = 0; r < 8; ++r) {
    int row = tM + r + half * 8;
    int col = tN + l16;
    float v = acc[r] + bv;
    if (relu) v = fmaxf(v, 0.0f);
    if (Cf) Cf[(size_t)row * Nn + col] = v;
    if (Cb) Cb[(size_t)row * Nn + col] = (bf16)v;
  }
}

// ---------------------------------------------------------------------------
// Fused cluster head: h = relu(x@W1+b1); logits = h@W2+b2; argmax ->
// atomic segment accumulation of f32 `current`.
// 4 waves / block, each wave owns a 16-row stripe.  The W1^T B-slab for the
// active 32-wide K window (32x256 bf16 = 16 KB) is staged into LDS once per
// block with double-buffered async global->LDS copies (ASYNCcnt), removing
// the 4x redundant per-wave global B traffic.
// ---------------------------------------------------------------------------
__global__ __launch_bounds__(128) void cluster_assign_accum_kernel(
    const bf16* __restrict__ curB, const float* __restrict__ curF,
    const int* __restrict__ labels, const bf16* __restrict__ w1T,  // [HH,DD]
    const float* __restrict__ b1, const bf16* __restrict__ w2T,    // [KC,HH]
    const float* __restrict__ b2, float* __restrict__ sums,
    float* __restrict__ cnt) {
  // slab[buf][n*32 + k] : B rows n=0..255, window-local k=0..31
  __shared__ __align__(16) bf16 slab[2][HH * 32];
  __shared__ __align__(16) bf16 ldsH[4][16][HH];
  __shared__ int ldsAssign[4][16];

  int tid = threadIdx.x;
  int wave = tid >> 5;
  int lane = tid & 31;
  int half = lane >> 4;
  int l16 = lane & 15;
  int tileM = blockIdx.x * 64 + wave * 16;

  // stage 16 KB B-slab (K window k0..k0+31) into slab[buf]: 128 threads x 8
  // chunks x 16 B, async (VGPR-bypassing), ASYNCcnt-tracked.
  auto stage = [&](int buf, int k0) {
#pragma unroll
    for (int c = 0; c < 8; ++c) {
      int i = tid + c * 128;  // 16B-chunk index, 0..1023 (4 chunks per B row)
      int n = i >> 2;
      int kc = i & 3;
      const bf16* g = w1T + (size_t)n * DD + k0 + kc * 8;
      unsigned int lo = (unsigned int)(uintptr_t)&slab[buf][n * 32 + kc * 8];
      async_copy_b128(lo, g);
    }
  };

  // ---- GEMM 1: [16,512] x [512,256] with 16 n-tile accumulators ----
  v8f acc[16];
#pragma unroll
  for (int i = 0; i < 16; ++i) acc[i] = (v8f)(0.0f);
  const bf16* aRow = curB + (size_t)(tileM + l16) * DD;

  stage(0, 0);
  for (int kidx = 0; kidx < DD / 32; ++kidx) {
    int k0 = kidx * 32;
    int buf = kidx & 1;
    wait_asynccnt0();    // our async writes for slab[buf] are done
    __syncthreads();     // everyone's are; prior iter's reads of slab[buf^1] done
    if (k0 + 32 < DD) stage(buf ^ 1, k0 + 32);

    Frag a;
    int ca = k0 + half * 8;
    a.u[0] = *(const q4*)(aRow + ca);
    a.u[1] = *(const q4*)(aRow + ca + 16);
    __builtin_prefetch(aRow + k0 + 128, 0, 0);
#pragma unroll
    for (int nt = 0; nt < 16; ++nt) {
      Frag b;
      const bf16* br = &slab[buf][(nt * 16 + l16) * 32 + half * 8];
      b.u[0] = *(const q4*)(br);
      b.u[1] = *(const q4*)(br + 16);
      acc[nt] = wmma_bf16(a, b, acc[nt]);
    }
  }

  // bias + relu -> LDS (row-major bf16, becomes A of GEMM 2)
#pragma unroll
  for (int nt = 0; nt < 16; ++nt) {
    int col = nt * 16 + l16;
    float bv = b1[col];
#pragma unroll
    for (int r = 0; r < 8; ++r) {
      int row = r + half * 8;
      float v = acc[nt][r] + bv;
      ldsH[wave][row][col] = (bf16)fmaxf(v, 0.0f);
    }
  }
  __syncthreads();

  // ---- GEMM 2: [16,256] x [256,16] logits ----
  v8f acc2 = {};
  for (int k0 = 0; k0 < HH; k0 += 32) {
    Frag a, b;
    int ca = k0 + half * 8;
    const bf16* lrow = &ldsH[wave][l16][0];
    a.u[0] = *(const q4*)(lrow + ca);
    a.u[1] = *(const q4*)(lrow + ca + 16);
    int cb = k0 + half * 8;
    const bf16* bRow = w2T + (size_t)l16 * HH + cb;
    b.u[0] = *(const q4*)(bRow);
    b.u[1] = *(const q4*)(bRow + 16);
    acc2 = wmma_bf16(a, b, acc2);
  }
  // argmax over 16 cluster columns (striped across the 16 lanes of each half)
  float bb = b2[l16];
#pragma unroll
  for (int r = 0; r < 8; ++r) {
    float v = acc2[r] + bb;
    int idx = l16;
#pragma unroll
    for (int off = 1; off < 16; off <<= 1) {
      float ov = __shfl_xor(v, off);
      int oi = __shfl_xor(idx, off);
      if (ov > v || (ov == v && oi < idx)) { v = ov; idx = oi; }
    }
    if (l16 == 0) ldsAssign[wave][r + half * 8] = idx;
  }
  __syncthreads();

  // ---- segment accumulation (f32 atomics into sums/cnt) ----
  for (int r = 0; r < 16; ++r) {
    int row = tileM + r;
    int seg = labels[row] * KC + ldsAssign[wave][r];
    const float* src = curF + (size_t)row * DD + lane * 16;
    float* dst = sums + (size_t)seg * DD + lane * 16;
#pragma unroll
    for (int j = 0; j < 16; ++j) atomicAdd(dst + j, src[j]);
    if (lane == 0) atomicAdd(cnt + seg, 1.0f);
  }
}

// ---------------------------------------------------------------------------
// Host-side orchestration
// ---------------------------------------------------------------------------
extern "C" void kernel_launch(void* const* d_in, const int* in_sizes, int n_in,
                              void* d_out, int out_size, void* d_ws,
                              size_t ws_size, hipStream_t stream) {
  const float* sf = (const float*)d_in[0];       // [N,D]
  const int* labels = (const int*)d_in[1];       // [N]
  const float* ch_w1 = (const float*)d_in[2];    // [L,D,H]
  const float* ch_b1 = (const float*)d_in[3];    // [L,H]
  const float* ch_w2 = (const float*)d_in[4];    // [L,H,K]
  const float* ch_b2 = (const float*)d_in[5];    // [L,K]
  const float* ref_w1 = (const float*)d_in[6];   // [L,D,D]
  const float* ref_b1 = (const float*)d_in[7];   // [L,D]
  const float* ln_g = (const float*)d_in[8];     // [L,D]
  const float* ln_b = (const float*)d_in[9];     // [L,D]
  const float* ref_w2 = (const float*)d_in[10];  // [L,D,D]
  const float* ref_b2 = (const float*)d_in[11];  // [L,D]
  const float* comb_w1 = (const float*)d_in[12]; // [L*D,D]
  const float* comb_b1 = (const float*)d_in[13]; // [D]
  const float* comb_w2 = (const float*)d_in[14]; // [D,D]
  const float* comb_b2 = (const float*)d_in[15]; // [D]
  float* out = (float*)d_out;                    // [C,D]

  char* ws = (char*)d_ws;
  size_t off = 0;
  auto carve = [&](size_t bytes) -> void* {
    void* p = ws + off;
    off += (bytes + 255) & ~(size_t)255;
    return p;
  };
  float* curF = (float*)carve((size_t)NN * DD * 4);
  bf16* curB = (bf16*)carve((size_t)NN * DD * 2);
  float* sums = (float*)carve((size_t)CK * DD * 4);
  float* cnt = (float*)carve((size_t)CK * 4);
  bf16* meansB = (bf16*)carve((size_t)CK * DD * 2);
  float* hpre = (float*)carve((size_t)CK * DD * 4);
  bf16* hB = (bf16*)carve((size_t)CK * DD * 2);
  float* refined = (float*)carve((size_t)CK * DD * 4);
  float* lpF = (float*)carve((size_t)CC * DD * 4);
  bf16* combB = (bf16*)carve((size_t)CC * LL * DD * 2);
  bf16* tB = (bf16*)carve((size_t)CC * DD * 2);
  bf16* w1T = (bf16*)carve((size_t)LL * HH * DD * 2);
  bf16* w2T = (bf16*)carve((size_t)LL * KC * HH * 2);
  bf16* r1T = (bf16*)carve((size_t)LL * DD * DD * 2);
  bf16* r2T = (bf16*)carve((size_t)LL * DD * DD * 2);
  bf16* c1T = (bf16*)carve((size_t)DD * (LL * DD) * 2);
  bf16* c2T = (bf16*)carve((size_t)DD * DD * 2);

  const int TB = 256;
  auto blocks = [&](int n) { return (n + TB - 1) / TB; };

  // current := support_features (f32 working copy + bf16 WMMA copy)
  init_current_kernel<<<blocks(NN * DD), TB, 0, stream>>>(sf, curF, curB, NN * DD);

  // weight transpose+bf16 conversion (per call; deterministic)
  for (int l = 0; l < LL; ++l) {
    transpose_bf16_kernel<<<blocks(DD * HH), TB, 0, stream>>>(
        ch_w1 + (size_t)l * DD * HH, w1T + (size_t)l * HH * DD, DD, HH);
    transpose_bf16_kernel<<<blocks(HH * KC), TB, 0, stream>>>(
        ch_w2 + (size_t)l * HH * KC, w2T + (size_t)l * KC * HH, HH, KC);
    transpose_bf16_kernel<<<blocks(DD * DD), TB, 0, stream>>>(
        ref_w1 + (size_t)l * DD * DD, r1T + (size_t)l * DD * DD, DD, DD);
    transpose_bf16_kernel<<<blocks(DD * DD), TB, 0, stream>>>(
        ref_w2 + (size_t)l * DD * DD, r2T + (size_t)l * DD * DD, DD, DD);
  }
  transpose_bf16_kernel<<<blocks(LL * DD * DD), TB, 0, stream>>>(comb_w1, c1T,
                                                                 LL * DD, DD);
  transpose_bf16_kernel<<<blocks(DD * DD), TB, 0, stream>>>(comb_w2, c2T, DD, DD);

  for (int l = 0; l < LL; ++l) {
    // zero segment sums + counts
    zero_f32_kernel<<<blocks(CK * DD), TB, 0, stream>>>(sums, CK * DD);
    zero_f32_kernel<<<blocks(CK), TB, 0, stream>>>(cnt, CK);

    // fused head GEMMs + argmax + segment accumulation
    cluster_assign_accum_kernel<<<NN / 64, 128, 0, stream>>>(
        curB, curF, labels, w1T + (size_t)l * HH * DD, ch_b1 + (size_t)l * HH,
        w2T + (size_t)l * KC * HH, ch_b2 + (size_t)l * KC, sums, cnt);

    // means (bf16 A operand for refiner)
    finalize_means_kernel<<<blocks(CK * DD), TB, 0, stream>>>(sums, cnt, meansB,
                                                              CK * DD);
    // refiner GEMM 1 (pre-LayerNorm, f32)
    wmma_gemm_kernel<<<(CK / 16) * (DD / 16), 32, 0, stream>>>(
        meansB, r1T + (size_t)l * DD * DD, ref_b1 + (size_t)l * DD, hpre,
        (bf16*)nullptr, CK, DD, DD, 0);
    // LayerNorm + ReLU -> bf16
    layernorm_relu_kernel<<<CK, 32, 0, stream>>>(hpre, ln_g + (size_t)l * DD,
                                                 ln_b + (size_t)l * DD, hB);
    // refiner GEMM 2 (f32 refined prototypes)
    wmma_gemm_kernel<<<(CK / 16) * (DD / 16), 32, 0, stream>>>(
        hB, r2T + (size_t)l * DD * DD, ref_b2 + (size_t)l * DD, refined,
        (bf16*)nullptr, CK, DD, DD, 0);
    // masked mean over non-empty clusters; also fills combined[:, l*D:(l+1)*D]
    level_proto_kernel<<<blocks(CC * DD), TB, 0, stream>>>(refined, cnt, lpF,
                                                           combB, l);
    // current += lp[label]  (skip on last level; result unused)
    if (l < LL - 1)
      update_current_kernel<<<blocks(NN * DD), TB, 0, stream>>>(curF, curB, lpF,
                                                                labels, NN * DD);
  }

  // combiner: relu(combined @ W1 + b1) @ W2 + b2
  wmma_gemm_kernel<<<(CC / 16) * (DD / 16), 32, 0, stream>>>(
      combB, c1T, comb_b1, (float*)nullptr, tB, CC, DD, LL * DD, 1);
  wmma_gemm_kernel<<<(CC / 16) * (DD / 16), 32, 0, stream>>>(
      tB, c2T, comb_b2, out, (bf16*)nullptr, CC, DD, DD, 0);
  (void)in_sizes; (void)n_in; (void)out_size; (void)ws_size;
}
